// MultiHeadAttention_85100482003265
// MI455X (gfx1250) — matrix operand order repaired and checked by reading, not compiled, not run on hardware
//
#include <hip/hip_runtime.h>

// ---- CDNA5 (gfx1250) wave32 WMMA types ----
typedef _Float16 v16h __attribute__((ext_vector_type(16)));
typedef _Float16 v8h  __attribute__((ext_vector_type(8)));
typedef float    v8f  __attribute__((ext_vector_type(8)));
typedef int      b128i __attribute__((ext_vector_type(4)));  // 16-byte block

constexpr int B_ = 4, S_ = 2048, E_ = 512, H_ = 8, D_ = 64;
constexpr int HD = H_ * D_; // 512

// CDNA5 async global->LDS path (ASYNCcnt), guarded so both toolchains
// compile. Probe-learned signature: (v4i AS(1)*, v4i AS(3)*, imm off, imm cpol).
#if defined(__has_builtin)
#  if __has_builtin(__builtin_amdgcn_global_load_async_to_lds_b128)
#    define ASYNC_LDS 1
#  endif
#endif
#ifndef ASYNC_LDS
#  define ASYNC_LDS 0
#endif

#if ASYNC_LDS
__device__ __forceinline__ void async_copy16(const _Float16* g, _Float16* l) {
  __builtin_amdgcn_global_load_async_to_lds_b128(
      (__attribute__((address_space(1))) b128i*)g,
      (__attribute__((address_space(3))) b128i*)l, 0, 0);
}
__device__ __forceinline__ void async_wait0() {
#  if __has_builtin(__builtin_amdgcn_s_wait_asynccnt)
  __builtin_amdgcn_s_wait_asynccnt(0);
#  else
  asm volatile("s_wait_asynccnt 0x0" ::: "memory");
#  endif
}
#endif

// ---------------------------------------------------------------------------
// f32 -> f16 conversion (x and weights), simple elementwise
// ---------------------------------------------------------------------------
__global__ void cvt_f32_f16(const float* __restrict__ src,
                            _Float16* __restrict__ dst, int n) {
  int i = blockIdx.x * blockDim.x + threadIdx.x;
  if (i < n) dst[i] = (_Float16)src[i];
}

// A-fragment loader: 16x32 f16 A-matrix layout (ISA 7.12.2).
// lane holds row M = lane%16; elements 0..7 = K in [hf*8, hf*8+8),
// elements 8..15 = K in [16+hf*8, 16+hf*8+8), hf = lane/16.
__device__ __forceinline__ v16h ld_a_frag(const _Float16* __restrict__ row,
                                          int kbase, int hf) {
  const _Float16* p = row + kbase + hf * 8;
  v8h lo = *(const v8h*)(p);
  v8h hi = *(const v8h*)(p + 16);
  v16h r;
#pragma unroll
  for (int i = 0; i < 8; ++i) { r[i] = lo[i]; r[i + 8] = hi[i]; }
  return r;
}

// ---------------------------------------------------------------------------
// QKV projection: [BS,E] x [E,HD] -> 16x64 tile per wave (one head's d-range).
// kt_layout=0: store out[b][h][s][d] (Q, V).  kt_layout=1: out[b][h][d][s] (K^T).
// ---------------------------------------------------------------------------
__global__ __launch_bounds__(32)
void qkv_kernel(const _Float16* __restrict__ X, const _Float16* __restrict__ W,
                const float* __restrict__ bias, _Float16* __restrict__ out,
                int kt_layout) {
  const int lane = threadIdx.x;
  const int m = lane & 15, hf = lane >> 4, n = lane & 15;
  const int it = blockIdx.x;   // 16-row tile of BS
  const int h  = blockIdx.y;   // head

  const _Float16* xrow = X + (size_t)(it * 16 + m) * E_;
  v8f acc[4] = {};

  for (int e0 = 0; e0 < E_; e0 += 32) {
    v16h xa = ld_a_frag(xrow, e0, hf);
#pragma unroll
    for (int t = 0; t < 4; ++t) {
      // B 32x16: lane holds column N = lane%16, elements = K in A's table order
      v16h wb;
#pragma unroll
      for (int s = 0; s < 16; ++s)
        wb[s] = W[(size_t)(e0 + (s & 7) + 8 * hf + 16 * (s >> 3)) * HD + h * 64 + 16 * t + n];
      acc[t] = __builtin_amdgcn_wmma_f32_16x16x32_f16(
          false, xa, false, wb, (short)0, acc[t], false, false);
    }
  }

#pragma unroll
  for (int t = 0; t < 4; ++t) {
    float bb = bias[h * 64 + 16 * t + n];
#pragma unroll
    for (int r = 0; r < 8; ++r) {
      int g = it * 16 + r + 8 * hf;        // global row in [0, B*S)
      int b = g / S_, s = g % S_;
      int d = 16 * t + n;
      float v = acc[t][r] + bb;
      size_t idx = kt_layout
          ? (((size_t)(b * H_ + h) * D_) + d) * S_ + s
          : (((size_t)(b * H_ + h) * S_) + s) * D_ + d;
      out[idx] = (_Float16)v;
    }
  }
}

// ---------------------------------------------------------------------------
// Fused flash attention. Block = 8 waves = 128 q rows of one (b,h).
// K^T/V chunks (64 k-columns) are staged into LDS once per block and shared
// by all 8 waves (8x less L2/HBM traffic than per-wave streaming).
// energy = QK^T/8 + rpe*rpe_c ; online softmax ; O += P*V ; O stored [B,S,H,D].
// ---------------------------------------------------------------------------
constexpr int WAVES = 8;
constexpr int QROWS = 16 * WAVES;  // 128 q rows per block

__global__ __launch_bounds__(32 * WAVES)
void attn_kernel(const _Float16* __restrict__ Q, const _Float16* __restrict__ Kt,
                 const _Float16* __restrict__ V, const float* __restrict__ rpe,
                 const float* __restrict__ rpe_c, _Float16* __restrict__ O) {
  __shared__ __align__(16) _Float16 ldsK[64 * 64];         // [d][k]  8KB
  __shared__ __align__(16) _Float16 ldsV[64 * 64];         // [k][d]  8KB
  __shared__ __align__(16) _Float16 ldsP[WAVES][16 * 64];  // P repack 16KB

  const int tid  = threadIdx.x;
  const int lane = tid & 31;
  const int wid  = tid >> 5;
  const int m = lane & 15, hf = lane >> 4, n = lane & 15;

  const int qb = blockIdx.x * QROWS + wid * 16;  // wave's q-row base
  const int h  = blockIdx.y;
  const int b  = blockIdx.z;
  const int bh = b * H_ + h;

  const _Float16* Qp = Q  + ((size_t)bh * S_ + qb) * D_;
  const _Float16* Kp = Kt + (size_t)bh * D_ * S_;   // [d][s]
  const _Float16* Vp = V  + (size_t)bh * S_ * D_;   // [s][d]
  const float*    rp = rpe + ((size_t)b * S_ + qb) * S_;
  const float rc = rpe_c[0];
  const float scale = 0.125f;  // 1/sqrt(64)

  // cooperative staging indices: 256 threads move 64x64 halfs (32B each)
  const int srow = tid >> 2;         // 0..63
  const int scol = (tid & 3) * 16;   // 0,16,32,48

  // Q A-fragments (d = 0..31, 32..63), loaded once
  v16h qa[2];
  qa[0] = ld_a_frag(Qp + m * D_, 0, hf);
  qa[1] = ld_a_frag(Qp + m * D_, 32, hf);

  float m_run[8], l_run[8], alpha[8];
#pragma unroll
  for (int r = 0; r < 8; ++r) { m_run[r] = -1e30f; l_run[r] = 0.0f; }
  v8f oacc[4] = {};

  for (int kc = 0; kc < S_; kc += 64) {
    // ---- stage K^T chunk [d][64] and V chunk [64][d] into LDS ----
    __syncthreads();  // all waves done reading previous chunk
#if ASYNC_LDS
    async_copy16(Kp + (size_t)srow * S_ + kc + scol,     &ldsK[srow * 64 + scol]);
    async_copy16(Kp + (size_t)srow * S_ + kc + scol + 8, &ldsK[srow * 64 + scol + 8]);
    async_copy16(Vp + (size_t)(kc + srow) * D_ + scol,     &ldsV[srow * 64 + scol]);
    async_copy16(Vp + (size_t)(kc + srow) * D_ + scol + 8, &ldsV[srow * 64 + scol + 8]);
    async_wait0();
#else
    {
      v16h kg = *(const v16h*)(Kp + (size_t)srow * S_ + kc + scol);
      *(v16h*)&ldsK[srow * 64 + scol] = kg;
      v16h vg = *(const v16h*)(Vp + (size_t)(kc + srow) * D_ + scol);
      *(v16h*)&ldsV[srow * 64 + scol] = vg;
    }
#endif
    __syncthreads();  // chunk visible to all waves

    // ---- energy tiles: 4 x (16x16), contraction d=64 = 2 WMMAs each ----
    v8f e[4] = {};
#pragma unroll
    for (int t = 0; t < 4; ++t) {
#pragma unroll
      for (int f = 0; f < 2; ++f) {
        v16h kb;
#pragma unroll
        for (int s = 0; s < 16; ++s)
          kb[s] = ldsK[(32 * f + (s & 7) + 8 * hf + 16 * (s >> 3)) * 64 + 16 * t + n];
        e[t] = __builtin_amdgcn_wmma_f32_16x16x32_f16(
            false, qa[f], false, kb, (short)0, e[t], false, false);
      }
    }

    // ---- scale + additive rpe bias ----
#pragma unroll
    for (int t = 0; t < 4; ++t) {
#pragma unroll
      for (int r = 0; r < 8; ++r) {
        int row = r + 8 * hf;
        e[t][r] = e[t][r] * scale + rp[(size_t)row * S_ + kc + 16 * t + n] * rc;
      }
    }

    // ---- online softmax: row max across 4 tiles + 16 lanes of half-wave ----
#pragma unroll
    for (int r = 0; r < 8; ++r) {
      float mx = e[0][r];
#pragma unroll
      for (int t = 1; t < 4; ++t) mx = fmaxf(mx, e[t][r]);
#pragma unroll
      for (int off = 1; off < 16; off <<= 1)
        mx = fmaxf(mx, __shfl_xor(mx, off, 32));
      float mn = fmaxf(m_run[r], mx);
      alpha[r] = __expf(m_run[r] - mn);
      m_run[r] = mn;
    }

    // ---- exp, row sums, spill P tile to this wave's LDS slab ----
#pragma unroll
    for (int r = 0; r < 8; ++r) {
      int row = r + 8 * hf;
      float rs = 0.0f;
#pragma unroll
      for (int t = 0; t < 4; ++t) {
        float pe = __expf(e[t][r] - m_run[r]);
        rs += pe;
        ldsP[wid][row * 64 + 16 * t + n] = (_Float16)pe;
      }
#pragma unroll
      for (int off = 1; off < 16; off <<= 1) rs += __shfl_xor(rs, off, 32);
      l_run[r] = l_run[r] * alpha[r] + rs;
    }

    // ---- rescale running output ----
#pragma unroll
    for (int od = 0; od < 4; ++od)
#pragma unroll
      for (int r = 0; r < 8; ++r) oacc[od][r] *= alpha[r];

    // ---- reload P as A-fragments (same-wave DS ops are in-order) ----
    v16h pa[2];
#pragma unroll
    for (int f = 0; f < 2; ++f) {
      const _Float16* p = &ldsP[wid][m * 64 + 32 * f + hf * 8];
      v8h lo = *(const v8h*)(p);
      v8h hi = *(const v8h*)(p + 16);
#pragma unroll
      for (int i = 0; i < 8; ++i) { pa[f][i] = lo[i]; pa[f][i + 8] = hi[i]; }
    }

    // ---- O += P (16x64) x V (64x64): 4 d-tiles x 2 k-frags ----
#pragma unroll
    for (int od = 0; od < 4; ++od) {
#pragma unroll
      for (int f = 0; f < 2; ++f) {
        v16h vb;
#pragma unroll
        for (int s = 0; s < 16; ++s)
          vb[s] = ldsV[(32 * f + (s & 7) + 8 * hf + 16 * (s >> 3)) * 64 + 16 * od + n];
        oacc[od] = __builtin_amdgcn_wmma_f32_16x16x32_f16(
            false, pa[f], false, vb, (short)0, oacc[od], false, false);
      }
    }
  }

  // ---- epilogue: normalize and store O as [B, S, H, D] f16 ----
#pragma unroll
  for (int od = 0; od < 4; ++od) {
#pragma unroll
    for (int r = 0; r < 8; ++r) {
      int row = r + 8 * hf;
      int s = qb + row;
      int d = 16 * od + n;
      float v = oacc[od][r] / l_run[r];
      O[(((size_t)b * S_ + s) * H_ + h) * D_ + d] = (_Float16)v;
    }
  }
}

// ---------------------------------------------------------------------------
// Output projection: O[BS, HD] x Wp[HD, D] + bp -> out f32 [BS, D]
// ---------------------------------------------------------------------------
__global__ __launch_bounds__(32)
void proj_kernel(const _Float16* __restrict__ Of, const _Float16* __restrict__ Wph,
                 const float* __restrict__ bp, float* __restrict__ out) {
  const int lane = threadIdx.x;
  const int m = lane & 15, hf = lane >> 4, n = lane & 15;
  const int it = blockIdx.x;

  const _Float16* orow = Of + (size_t)(it * 16 + m) * HD;
  v8f acc[4] = {};

  for (int e0 = 0; e0 < HD; e0 += 32) {
    v16h oa = ld_a_frag(orow, e0, hf);
#pragma unroll
    for (int t = 0; t < 4; ++t) {
      v16h wb;
#pragma unroll
      for (int s = 0; s < 16; ++s)
        wb[s] = Wph[(size_t)(e0 + (s & 7) + 8 * hf + 16 * (s >> 3)) * D_ + 16 * t + n];
      acc[t] = __builtin_amdgcn_wmma_f32_16x16x32_f16(
          false, oa, false, wb, (short)0, acc[t], false, false);
    }
  }

#pragma unroll
  for (int t = 0; t < 4; ++t) {
    float bb = bp[16 * t + n];
#pragma unroll
    for (int r = 0; r < 8; ++r) {
      int g = it * 16 + r + 8 * hf;
      out[(size_t)g * D_ + 16 * t + n] = acc[t][r] + bb;
    }
  }
}

// ---------------------------------------------------------------------------
// Host launcher
// ---------------------------------------------------------------------------
extern "C" void kernel_launch(void* const* d_in, const int* in_sizes, int n_in,
                              void* d_out, int out_size, void* d_ws, size_t ws_size,
                              hipStream_t stream) {
  (void)in_sizes; (void)n_in; (void)out_size; (void)ws_size;
  const float* x   = (const float*)d_in[0];
  const float* rpe = (const float*)d_in[1];
  const float* Wq  = (const float*)d_in[2];
  const float* bq  = (const float*)d_in[3];
  const float* Wk  = (const float*)d_in[4];
  const float* bk  = (const float*)d_in[5];
  const float* Wv  = (const float*)d_in[6];
  const float* bv  = (const float*)d_in[7];
  const float* Wp  = (const float*)d_in[8];
  const float* bp  = (const float*)d_in[9];
  const float* rpc = (const float*)d_in[10];
  float* out = (float*)d_out;

  // workspace carve-up (256B aligned)
  char* ws = (char*)d_ws;
  size_t off = 0;
  auto carve = [&](size_t bytes) -> char* {
    char* p = ws + off;
    off += (bytes + 255) & ~(size_t)255;
    return p;
  };
  _Float16* Xh  = (_Float16*)carve((size_t)B_ * S_ * E_ * 2);   // x f16
  _Float16* Wqh = (_Float16*)carve((size_t)E_ * HD * 2);
  _Float16* Wkh = (_Float16*)carve((size_t)E_ * HD * 2);
  _Float16* Wvh = (_Float16*)carve((size_t)E_ * HD * 2);
  _Float16* Wph = (_Float16*)carve((size_t)HD * D_ * 2);
  _Float16* Qh  = (_Float16*)carve((size_t)B_ * H_ * S_ * D_ * 2); // [b,h,s,d]
  _Float16* Kth = (_Float16*)carve((size_t)B_ * H_ * D_ * S_ * 2); // [b,h,d,s]
  _Float16* Vh  = (_Float16*)carve((size_t)B_ * H_ * S_ * D_ * 2); // [b,h,s,d]
  _Float16* Oh  = (_Float16*)carve((size_t)B_ * S_ * H_ * D_ * 2); // [b,s,h,d]

  // 0) convert inputs to f16
  {
    int n = B_ * S_ * E_;
    cvt_f32_f16<<<(n + 255) / 256, 256, 0, stream>>>(x, Xh, n);
    n = E_ * HD;
    cvt_f32_f16<<<(n + 255) / 256, 256, 0, stream>>>(Wq, Wqh, n);
    cvt_f32_f16<<<(n + 255) / 256, 256, 0, stream>>>(Wk, Wkh, n);
    cvt_f32_f16<<<(n + 255) / 256, 256, 0, stream>>>(Wv, Wvh, n);
    n = HD * D_;
    cvt_f32_f16<<<(n + 255) / 256, 256, 0, stream>>>(Wp, Wph, n);
  }

  // 1) QKV projections (K stored transposed for the QK^T B-operand)
  dim3 qkv_grid(B_ * S_ / 16, H_);
  qkv_kernel<<<qkv_grid, 32, 0, stream>>>(Xh, Wqh, bq, Qh, 0);
  qkv_kernel<<<qkv_grid, 32, 0, stream>>>(Xh, Wkh, bk, Kth, 1);
  qkv_kernel<<<qkv_grid, 32, 0, stream>>>(Xh, Wvh, bv, Vh, 0);

  // 2) fused flash attention with rpe bias (8 waves share K/V chunks via LDS)
  dim3 attn_grid(S_ / QROWS, H_, B_);
  attn_kernel<<<attn_grid, 32 * WAVES, 0, stream>>>(Qh, Kth, Vh, rpe, rpc, Oh);

  // 3) output projection
  proj_kernel<<<B_ * S_ / 16, 32, 0, stream>>>(Oh, Wph, bp, out);
}
